// GraphTransformerLayer_22660247454218
// MI455X (gfx1250) — matrix operand order repaired and checked by reading, not compiled, not run on hardware
//
#include <hip/hip_runtime.h>
#include <math.h>

typedef __attribute__((ext_vector_type(16))) __bf16 v16bf;
typedef __attribute__((ext_vector_type(8)))  __bf16 v8bf;
typedef __attribute__((ext_vector_type(8)))  float  v8f;

static constexpr int kB  = 8;
static constexpr int kN  = 128;
static constexpr int kXD = 256;
static constexpr int kED = 128;

// ---- CDNA5 async global->LDS staging (guarded; falls back to b128 ld/st) ----
#if defined(__HIP_DEVICE_COMPILE__) && defined(__has_builtin)
#if __has_builtin(__builtin_amdgcn_global_load_async_to_lds_b128) && \
    __has_builtin(__builtin_amdgcn_s_wait_asynccnt)
#define USE_ASYNC_LDS 1
#endif
#endif

#if defined(USE_ASYNC_LDS)
typedef int v4si __attribute__((__vector_size__(16)));
__device__ __forceinline__ void stage16B(const __bf16* g, __bf16* l) {
  typedef __attribute__((address_space(1))) v4si* g1p;   // global (AS1) int4*
  typedef __attribute__((address_space(3))) v4si* l3p;   // LDS (AS3) int4*
  // generic LDS address low 32 bits == LDS offset; global generic == AS1.
  __builtin_amdgcn_global_load_async_to_lds_b128(
      (g1p)(unsigned long long)(size_t)g, (l3p)(unsigned)(size_t)l, 0, 0);
}
__device__ __forceinline__ void stage_fence() {
  __builtin_amdgcn_s_wait_asynccnt(0);
}
#else
__device__ __forceinline__ void stage16B(const __bf16* g, __bf16* l) {
  *(v8bf*)l = *(const v8bf*)g;
}
__device__ __forceinline__ void stage_fence() {}
#endif

// ---------------------------------------------------------------- fp32->bf16
__global__ __launch_bounds__(256) void cvt_bf16_kernel(const float* __restrict__ in,
                                                       __bf16* __restrict__ out, int n) {
  int i = blockIdx.x * 256 + threadIdx.x;
  if (i < n) out[i] = (__bf16)in[i];
}

// fp32 W[K][N] -> bf16 W^T[N][K] (one-time; makes GEMM B tiles K-contiguous)
__global__ __launch_bounds__(256) void cvtT_bf16_kernel(const float* __restrict__ in,
                                                        __bf16* __restrict__ out,
                                                        int K, int N) {
  int i = blockIdx.x * 256 + threadIdx.x;
  if (i < K * N) {
    int k = i / N, n = i - k * N;
    out[(size_t)n * K + k] = (__bf16)in[i];
  }
}

// ---------------------------------------------------------------- generic bf16 WMMA GEMM
// C[M,N] = A[M,K] @ W[K,N] + bias (W given transposed: Wt[N][K]).
// Block = 128 threads (4 waves); block tile 64x64; wave tile 32x32 (2x2 WMMA); K-step 32.
__global__ __launch_bounds__(128) void gemm_bf16_kernel(
    const __bf16* __restrict__ A, const __bf16* __restrict__ Wt,
    const float* __restrict__ bias,
    float* __restrict__ outF, __bf16* __restrict__ outB,
    int M, int N, int K, int relu)
{
  __shared__ alignas(16) __bf16 As[64][40];
  __shared__ alignas(16) __bf16 Bs[64][40];
  const int tid  = threadIdx.x;
  const int lane = tid & 31;
  const int wave = tid >> 5;
  const int m0 = blockIdx.x * 64;
  const int n0 = blockIdx.y * 64;
  const int wm = (wave >> 1) * 32;
  const int wn = (wave & 1) * 32;
  const int lr = tid >> 2;          // 0..31
  const int lc = (tid & 3) * 8;     // 0,8,16,24

  v8f acc[2][2] = {};

  for (int k0 = 0; k0 < K; k0 += 32) {
    stage16B(A  + (size_t)(m0 + lr)      * K + k0 + lc, &As[lr][lc]);
    stage16B(A  + (size_t)(m0 + 32 + lr) * K + k0 + lc, &As[32 + lr][lc]);
    stage16B(Wt + (size_t)(n0 + lr)      * K + k0 + lc, &Bs[lr][lc]);
    stage16B(Wt + (size_t)(n0 + 32 + lr) * K + k0 + lc, &Bs[32 + lr][lc]);
    if (k0 + 32 < K)
      __builtin_prefetch(A + (size_t)(m0 + lr) * K + k0 + 32 + lc, 0, 1);
    stage_fence();
    __syncthreads();

    v16bf a[2], b[2];
    {
      const int ar = wm + (lane & 15);
      const int kb = (lane < 16) ? 0 : 16;
#pragma unroll
      for (int im = 0; im < 2; ++im) {
#pragma unroll
        for (int t = 0; t < 8; ++t) a[im][t]     = As[ar + im * 16][kb + t];
#pragma unroll
        for (int t = 0; t < 8; ++t) a[im][8 + t] = As[ar + im * 16][8 + kb + t];
      }
      const int br = wn + (lane & 15);
      const int ks = (lane < 16) ? 0 : 16;
#pragma unroll
      for (int in_ = 0; in_ < 2; ++in_) {
#pragma unroll
        for (int t = 0; t < 16; ++t) b[in_][t] = Bs[br + in_ * 16][ks + t];
      }
    }
#pragma unroll
    for (int im = 0; im < 2; ++im)
#pragma unroll
      for (int in_ = 0; in_ < 2; ++in_)
        acc[im][in_] = __builtin_amdgcn_wmma_f32_16x16x32_bf16(
            false, a[im], false, b[in_], (short)0, acc[im][in_], false, false);
    __syncthreads();
  }

#pragma unroll
  for (int im = 0; im < 2; ++im) {
    const int rb = m0 + wm + im * 16 + ((lane < 16) ? 0 : 8);
#pragma unroll
    for (int in_ = 0; in_ < 2; ++in_) {
      const int col = n0 + wn + in_ * 16 + (lane & 15);
      const float bv = bias ? bias[col] : 0.f;
#pragma unroll
      for (int r = 0; r < 8; ++r) {
        float v = acc[im][in_][r] + bv;
        if (relu) v = fmaxf(v, 0.f);
        size_t idx = (size_t)(rb + r) * N + col;
        if (outF) outF[idx] = v;
        if (outB) outB[idx] = (__bf16)v;
      }
    }
  }
}

// ---------------------------------------------------------------- fused edge GEMM
// Row r=(b,i,j): E1=e@wem+bem, E2=e@wea+bea via dual WMMA accumulators, then
// Y = (Q[b,i,c]*K[b,j,c]/sqrt(32))*(E1+1)+E2 ; emits Y fp32 (softmax) + bf16 (weo GEMM).
// M=131072, N=256(XD), K=128(ED). Weights given transposed [XD][ED].
__global__ __launch_bounds__(128) void edge_gemm_kernel(
    const __bf16* __restrict__ Eb,
    const __bf16* __restrict__ Wmt, const __bf16* __restrict__ Wat,
    const float* __restrict__ bem, const float* __restrict__ bea,
    const float* __restrict__ Qf, const float* __restrict__ Kf,
    float* __restrict__ Yf, __bf16* __restrict__ Yb)
{
  __shared__ alignas(16) __bf16 As[64][40];
  __shared__ alignas(16) __bf16 B1[64][40];
  __shared__ alignas(16) __bf16 B2[64][40];
  const int tid  = threadIdx.x;
  const int lane = tid & 31;
  const int wave = tid >> 5;
  const int m0 = blockIdx.x * 64;
  const int n0 = blockIdx.y * 64;
  const int wm = (wave >> 1) * 32;
  const int wn = (wave & 1) * 32;
  const int lr = tid >> 2;
  const int lc = (tid & 3) * 8;
  const int Kd = kED;

  v8f acc1[2][2] = {}, acc2[2][2] = {};

  for (int k0 = 0; k0 < Kd; k0 += 32) {
    stage16B(Eb  + (size_t)(m0 + lr)      * Kd + k0 + lc, &As[lr][lc]);
    stage16B(Eb  + (size_t)(m0 + 32 + lr) * Kd + k0 + lc, &As[32 + lr][lc]);
    stage16B(Wmt + (size_t)(n0 + lr)      * Kd + k0 + lc, &B1[lr][lc]);
    stage16B(Wmt + (size_t)(n0 + 32 + lr) * Kd + k0 + lc, &B1[32 + lr][lc]);
    stage16B(Wat + (size_t)(n0 + lr)      * Kd + k0 + lc, &B2[lr][lc]);
    stage16B(Wat + (size_t)(n0 + 32 + lr) * Kd + k0 + lc, &B2[32 + lr][lc]);
    stage_fence();
    __syncthreads();

    v16bf a[2], b1[2], b2[2];
    {
      const int ar = wm + (lane & 15);
      const int kb = (lane < 16) ? 0 : 16;
#pragma unroll
      for (int im = 0; im < 2; ++im) {
#pragma unroll
        for (int t = 0; t < 8; ++t) a[im][t]     = As[ar + im * 16][kb + t];
#pragma unroll
        for (int t = 0; t < 8; ++t) a[im][8 + t] = As[ar + im * 16][8 + kb + t];
      }
      const int br = wn + (lane & 15);
      const int ks = (lane < 16) ? 0 : 16;
#pragma unroll
      for (int in_ = 0; in_ < 2; ++in_) {
#pragma unroll
        for (int t = 0; t < 16; ++t) { b1[in_][t] = B1[br + in_ * 16][ks + t];
                                       b2[in_][t] = B2[br + in_ * 16][ks + t]; }
      }
    }
#pragma unroll
    for (int im = 0; im < 2; ++im)
#pragma unroll
      for (int in_ = 0; in_ < 2; ++in_) {
        acc1[im][in_] = __builtin_amdgcn_wmma_f32_16x16x32_bf16(
            false, a[im], false, b1[in_], (short)0, acc1[im][in_], false, false);
        acc2[im][in_] = __builtin_amdgcn_wmma_f32_16x16x32_bf16(
            false, a[im], false, b2[in_], (short)0, acc2[im][in_], false, false);
      }
    __syncthreads();
  }

  const float scale = 0.17677669529663687f;  // 1/sqrt(32)
#pragma unroll
  for (int im = 0; im < 2; ++im) {
    const int rg = m0 + wm + im * 16 + ((lane < 16) ? 0 : 8);
    const int j0 = rg & (kN - 1);
    const int bi = rg >> 7;   // b*128 + i
    const int bb = rg >> 14;  // b
#pragma unroll
    for (int in_ = 0; in_ < 2; ++in_) {
      const int col = n0 + wn + in_ * 16 + (lane & 15);
      const float q   = Qf[(size_t)bi * kXD + col];
      const float b1v = bem[col];
      const float b2v = bea[col];
      const size_t kidx = ((size_t)(bb * kN + j0)) * kXD + col;
#pragma unroll
      for (int r = 0; r < 8; ++r) {
        float kv = Kf[kidx + (size_t)r * kXD];
        float y  = (q * kv * scale) * (acc1[im][in_][r] + b1v + 1.0f)
                 + (acc2[im][in_][r] + b2v);
        size_t idx = (size_t)(rg + r) * kXD + col;
        Yf[idx] = y;
        Yb[idx] = (__bf16)y;
      }
    }
  }
}

// ---------------------------------------------------------------- online softmax over i + V reduce
__global__ __launch_bounds__(256) void softmax_wv_kernel(
    const float* __restrict__ Y, const float* __restrict__ V, float* __restrict__ wV)
{
  const int bj = blockIdx.x;
  const int c  = threadIdx.x;
  const int b  = bj >> 7, j = bj & (kN - 1);
  const float* yp = Y + ((size_t)b * kN * kN + j) * kXD + c;
  const float* vp = V + ((size_t)b * kN) * kXD + c;
  const size_t ystride = (size_t)kN * kXD;
  float m = -INFINITY, l = 0.f, acc = 0.f;
  for (int i = 0; i < kN; ++i) {
    float y  = yp[(size_t)i * ystride];
    float v  = vp[(size_t)i * kXD];
    float mn = fmaxf(m, y);
    float cr = __expf(m - mn);
    float p  = __expf(y - mn);
    l   = fmaf(l, cr, p);
    acc = fmaf(acc, cr, p * v);
    m = mn;
  }
  wV[(size_t)bj * kXD + c] = acc / l;
}

// ---------------------------------------------------------------- fused residual-add + LayerNorm
__global__ __launch_bounds__(256) void add_ln_kernel(
    const float* __restrict__ inp, const float* __restrict__ add,
    const float* __restrict__ g, const float* __restrict__ bn,
    __bf16* __restrict__ outB, float* __restrict__ outF, int D)
{
  __shared__ float red[8];
  __shared__ float s_mean, s_rstd;
  const int row = blockIdx.x, c = threadIdx.x;
  const int lane = c & 31, wv = c >> 5, nw = blockDim.x >> 5;
  const size_t idx = (size_t)row * D + c;
  float v = inp[idx] + add[idx];

  float s = v;
#pragma unroll
  for (int o = 16; o; o >>= 1) s += __shfl_xor(s, o, 32);
  if (lane == 0) red[wv] = s;
  __syncthreads();
  if (c == 0) {
    float t = 0.f;
    for (int w = 0; w < nw; ++w) t += red[w];
    s_mean = t / (float)D;
  }
  __syncthreads();
  const float mean = s_mean;
  const float d = v - mean;
  float q = d * d;
#pragma unroll
  for (int o = 16; o; o >>= 1) q += __shfl_xor(q, o, 32);
  if (lane == 0) red[wv] = q;
  __syncthreads();
  if (c == 0) {
    float t = 0.f;
    for (int w = 0; w < nw; ++w) t += red[w];
    s_rstd = rsqrtf(t / (float)D + 1e-5f);
  }
  __syncthreads();
  const float out = d * s_rstd * g[c] + bn[c];
  if (outB) outB[idx] = (__bf16)out;
  if (outF) outF[idx] = out;
}

// ================================================================ host
extern "C" void kernel_launch(void* const* d_in, const int* in_sizes, int n_in,
                              void* d_out, int out_size, void* d_ws, size_t ws_size,
                              hipStream_t stream)
{
  (void)in_sizes; (void)n_in; (void)out_size; (void)ws_size;

  const float* x   = (const float*)d_in[0];
  const float* e   = (const float*)d_in[1];
  const float* wq  = (const float*)d_in[2];
  const float* wk  = (const float*)d_in[3];
  const float* wv  = (const float*)d_in[4];
  const float* wem = (const float*)d_in[5];
  const float* wea = (const float*)d_in[6];
  const float* wxo = (const float*)d_in[7];
  const float* weo = (const float*)d_in[8];
  const float* wx1 = (const float*)d_in[9];
  const float* wx2 = (const float*)d_in[10];
  const float* we1 = (const float*)d_in[11];
  const float* we2 = (const float*)d_in[12];
  const float* bq  = (const float*)d_in[13];
  const float* bk  = (const float*)d_in[14];
  const float* bv  = (const float*)d_in[15];
  const float* bem = (const float*)d_in[16];
  const float* bea = (const float*)d_in[17];
  const float* bxo = (const float*)d_in[18];
  const float* beo = (const float*)d_in[19];
  const float* bx1 = (const float*)d_in[20];
  const float* bx2 = (const float*)d_in[21];
  const float* be1 = (const float*)d_in[22];
  const float* be2 = (const float*)d_in[23];
  const float* gnx = (const float*)d_in[24];
  const float* bnx = (const float*)d_in[25];
  const float* gne = (const float*)d_in[26];
  const float* bne = (const float*)d_in[27];

  const int Mx = kB * kN;             // 1024
  const int Me = kB * kN * kN;        // 131072

  char* base = (char*)d_ws;
  size_t off = 0;
  auto alloc = [&](size_t bytes) -> void* {
    void* p = base + off;
    off += (bytes + 255) & ~(size_t)255;
    return p;
  };

  __bf16* xb    = (__bf16*)alloc((size_t)Mx * kXD * 2);
  __bf16* ebf   = (__bf16*)alloc((size_t)Me * kED * 2);
  __bf16* wqt   = (__bf16*)alloc((size_t)kXD * kXD * 2);
  __bf16* wkt   = (__bf16*)alloc((size_t)kXD * kXD * 2);
  __bf16* wvt   = (__bf16*)alloc((size_t)kXD * kXD * 2);
  __bf16* wemt  = (__bf16*)alloc((size_t)kED * kXD * 2);
  __bf16* weat  = (__bf16*)alloc((size_t)kED * kXD * 2);
  __bf16* wxot  = (__bf16*)alloc((size_t)kXD * kXD * 2);
  __bf16* weot  = (__bf16*)alloc((size_t)kXD * kED * 2);
  __bf16* wx1t  = (__bf16*)alloc((size_t)kXD * 1024 * 2);
  __bf16* wx2t  = (__bf16*)alloc((size_t)1024 * kXD * 2);
  __bf16* we1t  = (__bf16*)alloc((size_t)kED * 512 * 2);
  __bf16* we2t  = (__bf16*)alloc((size_t)512 * kED * 2);
  float*  Qf    = (float*)alloc((size_t)Mx * kXD * 4);
  float*  Kf    = (float*)alloc((size_t)Mx * kXD * 4);
  float*  Vf    = (float*)alloc((size_t)Mx * kXD * 4);
  float*  Yf    = (float*)alloc((size_t)Me * kXD * 4);
  __bf16* Yb    = (__bf16*)alloc((size_t)Me * kXD * 2);
  float*  newEf = (float*)alloc((size_t)Me * kED * 4);
  float*  wVf   = (float*)alloc((size_t)Mx * kXD * 4);
  __bf16* wVb   = (__bf16*)alloc((size_t)Mx * kXD * 2);
  float*  newXf = (float*)alloc((size_t)Mx * kXD * 4);
  __bf16* t1b   = (__bf16*)alloc((size_t)Mx * kXD * 2);
  __bf16* h1b   = (__bf16*)alloc((size_t)Mx * 1024 * 2);
  float*  h2f   = (float*)alloc((size_t)Mx * kXD * 4);
  __bf16* teb   = (__bf16*)alloc((size_t)Me * kED * 2);
  __bf16* he1b  = (__bf16*)alloc((size_t)Me * 512 * 2);
  float*  he2f  = (float*)alloc((size_t)Me * kED * 4);

  auto cvt = [&](const float* in, __bf16* out, int n) {
    cvt_bf16_kernel<<<(n + 255) / 256, 256, 0, stream>>>(in, out, n);
  };
  auto cvtT = [&](const float* in, __bf16* out, int K, int N) {
    cvtT_bf16_kernel<<<(K * N + 255) / 256, 256, 0, stream>>>(in, out, K, N);
  };
  auto gemm = [&](const __bf16* A, const __bf16* Wt, const float* bias,
                  float* oF, __bf16* oB, int M, int N, int K, int relu) {
    dim3 grid(M / 64, N / 64);
    gemm_bf16_kernel<<<grid, 128, 0, stream>>>(A, Wt, bias, oF, oB, M, N, K, relu);
  };

  // ---- precision conversion (activations) + transpose-convert (weights)
  cvt(x, xb, Mx * kXD);
  cvt(e, ebf, Me * kED);
  cvtT(wq, wqt, kXD, kXD);   cvtT(wk, wkt, kXD, kXD);   cvtT(wv, wvt, kXD, kXD);
  cvtT(wem, wemt, kED, kXD); cvtT(wea, weat, kED, kXD);
  cvtT(wxo, wxot, kXD, kXD); cvtT(weo, weot, kXD, kED);
  cvtT(wx1, wx1t, kXD, 1024); cvtT(wx2, wx2t, 1024, kXD);
  cvtT(we1, we1t, kED, 512);  cvtT(we2, we2t, 512, kED);

  // ---- Q,K,V projections
  gemm(xb, wqt, bq, Qf, nullptr, Mx, kXD, kXD, 0);
  gemm(xb, wkt, bk, Kf, nullptr, Mx, kXD, kXD, 0);
  gemm(xb, wvt, bv, Vf, nullptr, Mx, kXD, kXD, 0);

  // ---- fused edge GEMMs + QK combine -> Y
  {
    dim3 grid(Me / 64, kXD / 64);
    edge_gemm_kernel<<<grid, 128, 0, stream>>>(ebf, wemt, weat, bem, bea, Qf, Kf, Yf, Yb);
  }

  // ---- newE = Y @ weo + beo
  gemm(Yb, weot, beo, newEf, nullptr, Me, kED, kXD, 0);

  // ---- attention: softmax over i + weighted V
  softmax_wv_kernel<<<kB * kN, kXD, 0, stream>>>(Yf, Vf, wVf);
  cvt(wVf, wVb, Mx * kXD);
  gemm(wVb, wxot, bxo, newXf, nullptr, Mx, kXD, kXD, 0);

  // ---- x feed-forward
  add_ln_kernel<<<Mx, kXD, 0, stream>>>(x, newXf, gnx, bnx, t1b, nullptr, kXD);
  gemm(t1b, wx1t, bx1, nullptr, h1b, Mx, 1024, kXD, 1);
  gemm(h1b, wx2t, bx2, h2f, nullptr, Mx, kXD, 1024, 0);
  float* outx = (float*)d_out;
  add_ln_kernel<<<Mx, kXD, 0, stream>>>(x, h2f, gnx, bnx, nullptr, outx, kXD);

  // ---- e feed-forward
  add_ln_kernel<<<Me, kED, 0, stream>>>(e, newEf, gne, bne, teb, nullptr, kED);
  gemm(teb, we1t, be1, nullptr, he1b, Me, 512, kED, 1);
  gemm(he1b, we2t, be2, he2f, nullptr, Me, kED, 512, 0);
  float* oute = outx + (size_t)Mx * kXD;
  add_ln_kernel<<<Me, kED, 0, stream>>>(e, he2f, gne, bne, nullptr, oute, kED);
}
